// ALLModel_73254962200757
// MI455X (gfx1250) — compile-verified, matrix operand order checked by reading, not run on hardware
//
#include <hip/hip_runtime.h>
#include <hip/hip_fp16.h>

// ---------------------------------------------------------------------------
// GCN (3x GCNConv + mean-pool + MLP head) for MI455X / gfx1250.
//
// Roofline: edge gather/scatter moves ~410MB/layer but the whole working set
// (~42MB) lives in the 192MB L2 -> L2/atomic bound, not HBM bound. Dense
// transforms (0.41 GFLOP each) go on v_wmma_f32_16x16x32_f16 (f32 accum).
// ---------------------------------------------------------------------------

#define NN   50000      // nodes
#define NE   800000     // edges
#define NG   512        // graphs
#define D    64         // hidden dim
#define DCLS 128

typedef __attribute__((ext_vector_type(16))) _Float16 v16h;
typedef __attribute__((ext_vector_type(8)))  float    v8f;

// ---------------------------------------------------------------------------
__global__ __launch_bounds__(256) void k_zero(float* __restrict__ p, int n) {
    int i = blockIdx.x * 256 + threadIdx.x;
    if (i < n) p[i] = 0.0f;
}

// deg[dst] += 1 (f32 atomics; deg buffer pre-zeroed)
__global__ __launch_bounds__(256) void k_degree(const int* __restrict__ dst,
                                                float* __restrict__ deg) {
    int e = blockIdx.x * 256 + threadIdx.x;
    if (e < NE) atomicAdd(&deg[dst[e]], 1.0f);
}

// in: deg (without self loop); out: deg_io = 1/(deg+1), dis = rsqrt(deg+1)
__global__ __launch_bounds__(256) void k_deg_finish(float* __restrict__ deg_io,
                                                    float* __restrict__ dis) {
    int n = blockIdx.x * 256 + threadIdx.x;
    if (n < NN) {
        float d = deg_io[n] + 1.0f;
        dis[n]    = rsqrtf(d);
        deg_io[n] = 1.0f / d;
    }
}

__global__ __launch_bounds__(256) void k_edge_norm(const int* __restrict__ src,
                                                   const int* __restrict__ dst,
                                                   const float* __restrict__ dis,
                                                   float* __restrict__ en) {
    int e = blockIdx.x * 256 + threadIdx.x;
    if (e < NE) en[e] = dis[src[e]] * dis[dst[e]];
}

// Transpose + convert a 64x64 f32 weight to f16: Wt[c*64+k] = (f16)W[k*64+c].
// Makes the WMMA B-fragment loads contiguous per lane.
__global__ __launch_bounds__(256) void k_prep_w(const float* __restrict__ W,
                                                _Float16* __restrict__ Wt) {
    int i = blockIdx.x * 256 + threadIdx.x;           // i = c*64 + k
    if (i < D * D) {
        int c = i >> 6, k = i & 63;
        Wt[i] = (_Float16)W[k * D + c];
    }
}

// ---------------------------------------------------------------------------
// hB[n,:] = A[n,:] @ W   (NNxD @ DxD), one wave per 16x16 output tile.
// A fragment (16x32 f16, ISA layout): lane L holds row M=L%16;
//   half e maps to K = (e/8)*16 + (e%8) + (L>=16 ? 8 : 0)  -> two contiguous
//   8-element chunks at [k0+hi8] and [k0+hi8+16]. B is symmetric with N=L%16
//   striped across lanes, reading the transposed f16 weight.
// C/D layout: VGPR r -> row r + (L>=16 ? 8:0), col = L%16.
// ---------------------------------------------------------------------------
__global__ __launch_bounds__(256) void k_gemm64(const float* __restrict__ A,
                                                const _Float16* __restrict__ Wt,
                                                float* __restrict__ Cout) {
    int wave = (blockIdx.x * 256 + threadIdx.x) >> 5;   // global wave id
    int lane = threadIdx.x & 31;
    const int tilesN = D / 16;                          // 4
    int tm = wave / tilesN;
    int tn = wave % tilesN;
    if (tm >= NN / 16) return;                          // wave-uniform exit

    int row = tm * 16 + (lane & 15);
    int col = tn * 16 + (lane & 15);
    int hi8 = (lane >> 4) << 3;                         // 0 or 8

    v8f acc = {};
#pragma unroll
    for (int k0 = 0; k0 < D; k0 += 32) {
        v16h a, b;
        const float*    ap = A  + row * D + k0 + hi8;
        const _Float16* bp = Wt + col * D + k0 + hi8;
#pragma unroll
        for (int e = 0; e < 8; ++e) {
            a[e]     = (_Float16)ap[e];
            a[e + 8] = (_Float16)ap[e + 16];
            b[e]     = bp[e];
            b[e + 8] = bp[e + 16];
        }
        acc = __builtin_amdgcn_wmma_f32_16x16x32_f16(
            /*neg_a=*/false, a, /*neg_b=*/false, b,
            /*c_mod=*/(short)0, acc, /*reuse_a=*/false, /*reuse_b=*/false);
    }
#pragma unroll
    for (int r = 0; r < 8; ++r)
        Cout[(tm * 16 + r + hi8) * D + col] = acc[r];
}

// ---------------------------------------------------------------------------
// agg[dst[e],:] += h[src[e],:] * en[e]. One wave per edge: the 256B source
// row is one coalesced gather; each lane owns a float2 column pair.
// ---------------------------------------------------------------------------
__global__ __launch_bounds__(256) void k_scatter(const float* __restrict__ h,
                                                 const int* __restrict__ src,
                                                 const int* __restrict__ dst,
                                                 const float* __restrict__ en,
                                                 float* __restrict__ agg) {
    int gid = blockIdx.x * 256 + threadIdx.x;
    int e = gid >> 5;
    if (e >= NE) return;                                // wave-uniform
    int lane = threadIdx.x & 31;
    int s = src[e], d = dst[e];
    float nrm = en[e];
    const float2 v = *(const float2*)(h + (size_t)s * D + lane * 2);
    float* o = agg + (size_t)d * D + lane * 2;
    atomicAdd(o,     v.x * nrm);
    atomicAdd(o + 1, v.y * nrm);
}

// out = agg + deg_inv[n]*hW + b  (self-loop term + bias)
__global__ __launch_bounds__(256) void k_combine(const float* __restrict__ agg,
                                                 const float* __restrict__ hW,
                                                 const float* __restrict__ dinv,
                                                 const float* __restrict__ b,
                                                 float* __restrict__ out) {
    int i = blockIdx.x * 256 + threadIdx.x;
    if (i < NN * D) {
        int n = i >> 6, c = i & 63;
        out[i] = agg[i] + dinv[n] * hW[i] + b[c];
    }
}

// mean-pool accumulation: sums[batch[n],:] += h[n,:]; counts[batch[n]] += 1
__global__ __launch_bounds__(256) void k_pool(const float* __restrict__ h,
                                              const int* __restrict__ batch,
                                              float* __restrict__ sums,
                                              float* __restrict__ counts) {
    int gid = blockIdx.x * 256 + threadIdx.x;
    int n = gid >> 5;
    if (n >= NN) return;
    int lane = threadIdx.x & 31;
    int g = batch[n];
    const float2 v = *(const float2*)(h + (size_t)n * D + lane * 2);
    float* o = sums + (size_t)g * D + lane * 2;
    atomicAdd(o,     v.x);
    atomicAdd(o + 1, v.y);
    if (lane == 0) atomicAdd(&counts[g], 1.0f);
}

// head: pooled = sums/max(counts,1); (64->128)@lin1 + b; @lin2 + b; *fc + b;
// LeakyReLU(0.01). One 128-thread block per graph (512 blocks, trivial work).
__global__ __launch_bounds__(128) void k_head(const float* __restrict__ sums,
                                              const float* __restrict__ counts,
                                              const float* __restrict__ l1w,
                                              const float* __restrict__ l1b,
                                              const float* __restrict__ l2w,
                                              const float* __restrict__ l2b,
                                              const float* __restrict__ fcw,
                                              const float* __restrict__ fcb,
                                              float* __restrict__ out) {
    __shared__ float pooled[D];
    __shared__ float red[DCLS];
    int g = blockIdx.x;
    int t = threadIdx.x;
    float cnt = fmaxf(counts[g], 1.0f);
    if (t < D) pooled[t] = sums[g * D + t] / cnt;
    __syncthreads();

    float acc = l1b[t];
#pragma unroll
    for (int k = 0; k < D; ++k)
        acc = fmaf(pooled[k], l1w[k * DCLS + t], acc);
    red[t] = acc * l2w[t];
    __syncthreads();
    for (int s = DCLS / 2; s > 0; s >>= 1) {
        if (t < s) red[t] += red[t + s];
        __syncthreads();
    }
    if (t == 0) {
        float o = red[0] + l2b[0];
        o = o * fcw[0] + fcb[0];
        out[g] = (o >= 0.0f) ? o : 0.01f * o;
    }
}

// ---------------------------------------------------------------------------
static inline int cdiv(int a, int b) { return (a + b - 1) / b; }

extern "C" void kernel_launch(void* const* d_in, const int* in_sizes, int n_in,
                              void* d_out, int out_size, void* d_ws, size_t ws_size,
                              hipStream_t stream) {
    (void)in_sizes; (void)n_in; (void)out_size; (void)ws_size;

    const float* x   = (const float*)d_in[0];
    const float* W0  = (const float*)d_in[1];
    const float* b0  = (const float*)d_in[2];
    const float* W1  = (const float*)d_in[3];
    const float* b1  = (const float*)d_in[4];
    const float* W2  = (const float*)d_in[5];
    const float* b2  = (const float*)d_in[6];
    const float* l1w = (const float*)d_in[7];
    const float* l1b = (const float*)d_in[8];
    const float* l2w = (const float*)d_in[9];
    const float* l2b = (const float*)d_in[10];
    const float* fcw = (const float*)d_in[11];
    const float* fcb = (const float*)d_in[12];
    const int*   ei  = (const int*)d_in[13];
    const int*   bat = (const int*)d_in[14];
    const int* src = ei;            // edge_index[0,:]
    const int* dst = ei + NE;       // edge_index[1,:]
    float* out = (float*)d_out;

    // workspace partition (~42.2 MB, fits comfortably in L2)
    float* ws     = (float*)d_ws;
    float* dis    = ws;  ws += NN;          // deg_inv_sqrt
    float* dinv   = ws;  ws += NN;          // deg accum -> deg_inv
    float* en     = ws;  ws += NE;          // edge_norm
    float* hA     = ws;  ws += NN * D;      // current features
    float* hB     = ws;  ws += NN * D;      // h @ W
    float* hC     = ws;  ws += NN * D;      // scatter accumulator
    float* sums   = ws;  ws += NG * D;      // pooled sums
    float* counts = ws;  ws += NG;          // pooled counts (contiguous w/ sums)
    _Float16* Wt0 = (_Float16*)ws;
    _Float16* Wt1 = Wt0 + D * D;
    _Float16* Wt2 = Wt1 + D * D;

    // --- degrees / norms (computed once, reused by all 3 layers) ---
    k_zero      <<<cdiv(NN, 256), 256, 0, stream>>>(dinv, NN);
    k_degree    <<<cdiv(NE, 256), 256, 0, stream>>>(dst, dinv);
    k_deg_finish<<<cdiv(NN, 256), 256, 0, stream>>>(dinv, dis);
    k_edge_norm <<<cdiv(NE, 256), 256, 0, stream>>>(src, dst, dis, en);

    // --- f16 transposed weights for WMMA B fragments ---
    k_prep_w<<<cdiv(D * D, 256), 256, 0, stream>>>(W0, Wt0);
    k_prep_w<<<cdiv(D * D, 256), 256, 0, stream>>>(W1, Wt1);
    k_prep_w<<<cdiv(D * D, 256), 256, 0, stream>>>(W2, Wt2);

    // --- 3 GCN layers ---
    const int TILES = (NN / 16) * (D / 16);             // 3125 * 4 = 12500 waves
    const float*    Hin[3]  = { x, hA, hA };
    const _Float16* Wt[3]   = { Wt0, Wt1, Wt2 };
    const float*    bias[3] = { b0, b1, b2 };
    for (int L = 0; L < 3; ++L) {
        k_gemm64 <<<cdiv(TILES * 32, 256), 256, 0, stream>>>(Hin[L], Wt[L], hB);
        k_zero   <<<cdiv(NN * D, 256), 256, 0, stream>>>(hC, NN * D);
        k_scatter<<<cdiv(NE * 32, 256), 256, 0, stream>>>(hB, src, dst, en, hC);
        k_combine<<<cdiv(NN * D, 256), 256, 0, stream>>>(hC, hB, dinv, bias[L], hA);
    }

    // --- mean pool + head ---
    k_zero<<<cdiv(NG * D + NG, 256), 256, 0, stream>>>(sums, NG * D + NG);
    k_pool<<<cdiv(NN * 32, 256), 256, 0, stream>>>(hA, bat, sums, counts);
    k_head<<<NG, 128, 0, stream>>>(sums, counts, l1w, l1b, l2w, l2b, fcw, fcb, out);
}
